// AttentionBlock_23106924052848
// MI455X (gfx1250) — matrix operand order repaired and checked by reading, not compiled, not run on hardware
//
#include <hip/hip_runtime.h>
#include <math.h>

// ---- problem constants (B=2, C=64, H=W=L=32, heads=4, d=16) ----
#define TOK_TOTAL 65536   // B*H*W*L tokens
#define CCH 64

typedef _Float16 half_t;
typedef _Float16 v16h __attribute__((ext_vector_type(16)));
typedef float    v8f  __attribute__((ext_vector_type(8)));
typedef int      v4i  __attribute__((ext_vector_type(4)));

union HV { struct { v4i lo, hi; } q; v16h h; };

__device__ __forceinline__ v16h make16(v4i lo, v4i hi) {
  HV u; u.q.lo = lo; u.q.hi = hi; return u.h;
}
__device__ __forceinline__ v4i ld4i(const half_t* p) {
  return *(const v4i*)p;
}
__device__ __forceinline__ v8f wmma16(v16h a, v16h b, v8f c) {
  // D = A(16x32 f16) x B(32x16 f16) + C(16x16 f32)
  return __builtin_amdgcn_wmma_f32_16x16x32_f16(false, a, false, b, (short)0, c, false, false);
}

// ---------------- GroupNorm ----------------
__global__ void gn_stats_kernel(const float* __restrict__ x, float* __restrict__ stats) {
  __shared__ float s1[256], s2[256];
  int gid = blockIdx.x;                 // 0..15 : (b, group)
  int b = gid >> 3, g = gid & 7;
  const float* base = x + ((size_t)(b * 64 + g * 8) << 15);   // 8 channels * 32768 contiguous
  float sum = 0.f, ss = 0.f;
  for (int i = threadIdx.x; i < (8 << 15); i += 256) {
    float v = base[i];
    sum += v; ss += v * v;
  }
  s1[threadIdx.x] = sum; s2[threadIdx.x] = ss;
  __syncthreads();
  for (int st = 128; st > 0; st >>= 1) {
    if ((int)threadIdx.x < st) {
      s1[threadIdx.x] += s1[threadIdx.x + st];
      s2[threadIdx.x] += s2[threadIdx.x + st];
    }
    __syncthreads();
  }
  if (threadIdx.x == 0) {
    float n = 8.0f * 32768.0f;
    float mu = s1[0] / n;
    float var = s2[0] / n - mu * mu;
    stats[gid]      = mu;
    stats[16 + gid] = rsqrtf(var + 1e-5f);
  }
}

// apply GN and write fp16 tokens in spatial order: tok[(b*L+l)*HW + (h*W+w)][c]
__global__ void gn_apply_kernel(const float* __restrict__ x, const float* __restrict__ gamma,
                                const float* __restrict__ beta, const float* __restrict__ stats,
                                half_t* __restrict__ tok) {
  int f = blockIdx.x * 256 + threadIdx.x;     // < 4194304, x flat index [b,c,h,w,l]
  int l = f & 31, w = (f >> 5) & 31, h = (f >> 10) & 31, c = (f >> 15) & 63, b = f >> 21;
  int sg = b * 8 + (c >> 3);
  float v = (x[f] - stats[sg]) * stats[16 + sg] * gamma[c] + beta[c];
  int tokid = ((b << 5) + l) * 1024 + (h << 5) + w;
  tok[(size_t)tokid * 64 + c] = (half_t)v;
}

// ---------------- weight fp32 -> fp16 ----------------
__global__ void pack_w_kernel(const float* __restrict__ w, half_t* __restrict__ wh, int n) {
  int i = blockIdx.x * 256 + threadIdx.x;
  if (i < n) wh[i] = (half_t)w[i];
}

// ---------------- QKV projection GEMM ----------------
// tok [T,64] fp16 @ Wh[192,64]^T + bias -> Q [t,64]*0.25, K [t,64], V transposed [seq][c][S]
// sShift = log2(S)
__global__ void qkv_gemm_kernel(const half_t* __restrict__ tok, const half_t* __restrict__ Wh,
                                const float* __restrict__ bias,
                                half_t* __restrict__ Qb, half_t* __restrict__ Kb,
                                half_t* __restrict__ Vt, int sShift) {
  int lane = threadIdx.x & 31;
  int wid  = blockIdx.x * 4 + (threadIdx.x >> 5);
  int mt = wid / 12, nt = wid % 12;            // /12: strength-reduced constant division
  int row0 = mt * 16;
  int ln16 = lane & 15;
  int kbA = (lane < 16) ? 0 : 8;
  int kbB = (lane < 16) ? 0 : 8;
  const half_t* arow = tok + (size_t)(row0 + ln16) * 64 + kbA;
  const half_t* brow = Wh  + (size_t)(nt * 16 + ln16) * 64 + kbB;
  __builtin_prefetch(brow, 0, 3);              // global_prefetch_b8: weight tile is hot
  v8f acc = {0.f,0.f,0.f,0.f,0.f,0.f,0.f,0.f};
#pragma unroll
  for (int k0 = 0; k0 < 64; k0 += 32) {
    v16h a = make16(ld4i(arow + k0), ld4i(arow + k0 + 16));
    v16h b = make16(ld4i(brow + k0), ld4i(brow + k0 + 16));
    acc = wmma16(a, b, acc);
  }
  int n = nt * 16 + ln16;
  float bv = bias[n];
  int rbase = (lane < 16) ? 0 : 8;
  int sMask = (1 << sShift) - 1;
#pragma unroll
  for (int i = 0; i < 8; i++) {
    int t = row0 + rbase + i;
    float val = acc[i] + bv;
    if (n < 64) {
      Qb[(size_t)t * 64 + n] = (half_t)(val * 0.25f);           // fold 1/sqrt(d)
    } else if (n < 128) {
      Kb[(size_t)t * 64 + (n - 64)] = (half_t)val;
    } else {
      int sq = t >> sShift, s = t & sMask;
      Vt[(((size_t)sq * 64 + (n - 128)) << sShift) + s] = (half_t)val;  // V^T layout
    }
  }
}

// ---------------- flash attention ----------------
// one wave per (seq, head, 16-row query tile); sShift = log2(S), S in {1024, 32}
__global__ void flash_attn_kernel(const half_t* __restrict__ Qb, const half_t* __restrict__ Kb,
                                  const half_t* __restrict__ Vt, half_t* __restrict__ O,
                                  int sShift) {
  __shared__ __align__(16) half_t Pl[4][16 * 32];
  int S = 1 << sShift;
  int lane = threadIdx.x & 31;
  int wslot = threadIdx.x >> 5;
  int wid = blockIdx.x * 4 + wslot;
  int hd = wid & 3;
  int qg = wid >> 2;                  // 0..4095
  int seq = qg >> (sShift - 4);       // qPerSeq = S/16
  int qt  = qg & ((1 << (sShift - 4)) - 1);
  int m = lane & 15;
  int kbA = (lane < 16) ? 0 : 8;
  int rbase = (lane < 16) ? 0 : 8;
  v4i z4 = {0, 0, 0, 0};

  // Q tile as A operand, K-dim padded to 32 with zeros (d=16)
  const half_t* qp = Qb + (size_t)(seq * S + qt * 16 + m) * 64 + hd * 16;
  v16h aq = (lane < 16) ? make16(ld4i(qp), ld4i(qp + 8)) : make16(z4, z4);

  const half_t* kbase = Kb + (size_t)seq * S * 64 + hd * 16;
  const half_t* vbase = Vt + (((size_t)seq * 64 + hd * 16 + m) << sShift) + ((lane < 16) ? 0 : 8);
  half_t* pl = Pl[wslot];

  v8f oc = {0.f,0.f,0.f,0.f,0.f,0.f,0.f,0.f};
  float mr[8], lr[8];
#pragma unroll
  for (int i = 0; i < 8; i++) { mr[i] = -1e30f; lr[i] = 0.f; }

  int nkb = S >> 5;
  for (int kb = 0; kb < nkb; kb++) {
    v16h b0, b1;
    if (lane < 16) {   // lanes 0..15 hold K(d)=0..15 rows; d>=16 rows are zero padding
      const half_t* kp0 = kbase + (size_t)(kb * 32 + m) * 64;
      const half_t* kp1 = kbase + (size_t)(kb * 32 + 16 + m) * 64;
      b0 = make16(ld4i(kp0), ld4i(kp0 + 8));
      b1 = make16(ld4i(kp1), ld4i(kp1 + 8));
    } else {
      b0 = make16(z4, z4);
      b1 = make16(z4, z4);
    }
    v8f zz = {0.f,0.f,0.f,0.f,0.f,0.f,0.f,0.f};
    v8f s0 = wmma16(aq, b0, zz);
    v8f s1 = wmma16(aq, b1, zz);
#pragma unroll
    for (int i = 0; i < 8; i++) {
      float tmax = fmaxf(s0[i], s1[i]);
      tmax = fmaxf(tmax, __shfl_xor(tmax, 1, 32));
      tmax = fmaxf(tmax, __shfl_xor(tmax, 2, 32));
      tmax = fmaxf(tmax, __shfl_xor(tmax, 4, 32));
      tmax = fmaxf(tmax, __shfl_xor(tmax, 8, 32));
      float mn = fmaxf(mr[i], tmax);
      float al = __expf(mr[i] - mn);
      float p0 = __expf(s0[i] - mn);
      float p1 = __expf(s1[i] - mn);
      float rs = p0 + p1;
      rs += __shfl_xor(rs, 1, 32);
      rs += __shfl_xor(rs, 2, 32);
      rs += __shfl_xor(rs, 4, 32);
      rs += __shfl_xor(rs, 8, 32);
      lr[i] = lr[i] * al + rs;
      mr[i] = mn;
      oc[i] = oc[i] * al;
      pl[(rbase + i) * 32 + m]      = (half_t)p0;   // D-layout -> LDS
      pl[(rbase + i) * 32 + 16 + m] = (half_t)p1;
    }
    // same-wave LDS ops are in-order; fence the compiler + drain DS before A-layout reads
    __asm__ volatile("s_wait_dscnt 0x0" ::: "memory");
    v16h ap = make16(ld4i(pl + m * 32 + kbA), ld4i(pl + m * 32 + 16 + kbA));
    const half_t* vp = vbase + kb * 32;
    v16h bv = make16(ld4i(vp), ld4i(vp + 16));
    oc = wmma16(ap, bv, oc);
    __asm__ volatile("" ::: "memory");
  }
  int t0 = seq * S + qt * 16 + rbase;
#pragma unroll
  for (int i = 0; i < 8; i++) {
    float v = oc[i] / lr[i];
    O[(size_t)(t0 + i) * 64 + hd * 16 + m] = (half_t)v;
  }
}

// ---------------- output projection GEMM (optionally fused residual + permute) ----------------
__global__ void out_gemm_kernel(const half_t* __restrict__ A, const half_t* __restrict__ Wh,
                                const float* __restrict__ bias, half_t* __restrict__ out16,
                                const float* __restrict__ xres, float* __restrict__ outf,
                                int resid) {
  int lane = threadIdx.x & 31;
  int wid = blockIdx.x * 4 + (threadIdx.x >> 5);
  int mt = wid >> 2, nt = wid & 3;
  int row0 = mt * 16;
  int ln16 = lane & 15;
  int kbA = (lane < 16) ? 0 : 8;
  int kbB = (lane < 16) ? 0 : 8;
  const half_t* arow = A  + (size_t)(row0 + ln16) * 64 + kbA;
  const half_t* brow = Wh + (size_t)(nt * 16 + ln16) * 64 + kbB;
  __builtin_prefetch(brow, 0, 3);
  v8f acc = {0.f,0.f,0.f,0.f,0.f,0.f,0.f,0.f};
#pragma unroll
  for (int k0 = 0; k0 < 64; k0 += 32) {
    v16h a = make16(ld4i(arow + k0), ld4i(arow + k0 + 16));
    v16h b = make16(ld4i(brow + k0), ld4i(brow + k0 + 16));
    acc = wmma16(a, b, acc);
  }
  int n = nt * 16 + ln16;
  float bv = bias[n];
  int rbase = (lane < 16) ? 0 : 8;
#pragma unroll
  for (int i = 0; i < 8; i++) {
    int t = row0 + rbase + i;
    float val = acc[i] + bv;
    if (!resid) {
      out16[(size_t)t * 64 + n] = (half_t)val;
    } else {
      // spatial token t = (b*32+l)*1024 + (h*32+w); x flat = (((b*64+c)*32+h)*32+w)*32+l
      int b = t >> 15, l = (t >> 10) & 31, hh = (t >> 5) & 31, ww = t & 31;
      size_t flat = ((((size_t)b * 64 + n) * 32 + hh) * 32 + ww) * 32 + l;
      outf[flat] = xres[flat] + val;
    }
  }
}

// ---------------- token layout permutation (spatial <-> temporal order) ----------------
__global__ void relayout_kernel(const half_t* __restrict__ src, half_t* __restrict__ dst, int s2t) {
  int id = blockIdx.x * 256 + threadIdx.x;   // 524288 threads, 8 halfs each
  int tt = id >> 3;                          // destination token
  int seg = id & 7;
  int b = tt >> 15;
  int r = tt & 32767;
  int ts;
  if (s2t) {              // dst temporal: tt = (b*1024 + p)*32 + l
    int p = r >> 5, l = r & 31;
    ts = ((b << 5) + l) * 1024 + p;          // src spatial
  } else {                // dst spatial: tt = (b*32 + l)*1024 + p
    int l = r >> 10, p = r & 1023;
    ts = ((b << 10) + p) * 32 + l;           // src temporal
  }
  *(v4i*)(dst + (size_t)tt * 64 + seg * 8) =
      *(const v4i*)(src + (size_t)ts * 64 + seg * 8);
}

extern "C" void kernel_launch(void* const* d_in, const int* in_sizes, int n_in,
                              void* d_out, int out_size, void* d_ws, size_t ws_size,
                              hipStream_t stream) {
  (void)in_sizes; (void)n_in; (void)out_size; (void)ws_size;
  const float* x         = (const float*)d_in[0];
  const float* gn_gamma  = (const float*)d_in[1];
  const float* gn_beta   = (const float*)d_in[2];
  const float* spa_in_w  = (const float*)d_in[3];
  const float* spa_in_b  = (const float*)d_in[4];
  const float* spa_out_w = (const float*)d_in[5];
  const float* spa_out_b = (const float*)d_in[6];
  const float* tem_in_w  = (const float*)d_in[7];
  const float* tem_in_b  = (const float*)d_in[8];
  const float* tem_out_w = (const float*)d_in[9];
  const float* tem_out_b = (const float*)d_in[10];
  float* outp = (float*)d_out;

  char* ws = (char*)d_ws;
  size_t off = 0;
  auto alloc = [&](size_t bytes) -> void* {
    void* p = ws + off;
    off = (off + bytes + 255) & ~(size_t)255;
    return p;
  };
  float*  stats = (float*)alloc(32 * sizeof(float));
  half_t* w_si  = (half_t*)alloc(12288 * sizeof(half_t));
  half_t* w_so  = (half_t*)alloc(4096 * sizeof(half_t));
  half_t* w_ti  = (half_t*)alloc(12288 * sizeof(half_t));
  half_t* w_to  = (half_t*)alloc(4096 * sizeof(half_t));
  size_t tokB = (size_t)TOK_TOTAL * 64 * sizeof(half_t);   // 8 MB each
  half_t* tok0 = (half_t*)alloc(tokB);
  half_t* tok1 = (half_t*)alloc(tokB);
  half_t* Qb   = (half_t*)alloc(tokB);
  half_t* Kb   = (half_t*)alloc(tokB);
  half_t* Vt   = (half_t*)alloc(tokB);
  half_t* attn = (half_t*)alloc(tokB);

  // weights -> fp16
  pack_w_kernel<<<48, 256, 0, stream>>>(spa_in_w,  w_si, 12288);
  pack_w_kernel<<<16, 256, 0, stream>>>(spa_out_w, w_so, 4096);
  pack_w_kernel<<<48, 256, 0, stream>>>(tem_in_w,  w_ti, 12288);
  pack_w_kernel<<<16, 256, 0, stream>>>(tem_out_w, w_to, 4096);

  // group norm -> spatial-order fp16 tokens
  gn_stats_kernel<<<16, 256, 0, stream>>>(x, stats);
  gn_apply_kernel<<<16384, 256, 0, stream>>>(x, gn_gamma, gn_beta, stats, tok0);

  // pass 1: spatial MHSA (S=1024, log2=10), result fp16 in tok1 (spatial order)
  qkv_gemm_kernel<<<12288, 128, 0, stream>>>(tok0, w_si, spa_in_b, Qb, Kb, Vt, 10);
  flash_attn_kernel<<<4096, 128, 0, stream>>>(Qb, Kb, Vt, attn, 10);
  out_gemm_kernel<<<4096, 128, 0, stream>>>(attn, w_so, spa_out_b, tok1, nullptr, nullptr, 0);
  relayout_kernel<<<2048, 256, 0, stream>>>(tok1, tok0, 1);   // -> temporal order

  // pass 2: temporal MHSA (S=32, log2=5)
  qkv_gemm_kernel<<<12288, 128, 0, stream>>>(tok0, w_ti, tem_in_b, Qb, Kb, Vt, 5);
  flash_attn_kernel<<<4096, 128, 0, stream>>>(Qb, Kb, Vt, attn, 5);
  out_gemm_kernel<<<4096, 128, 0, stream>>>(attn, w_to, tem_out_b, tok1, nullptr, nullptr, 0);
  relayout_kernel<<<2048, 256, 0, stream>>>(tok1, tok0, 0);   // -> spatial order

  // pass 3: spatial MHSA (S=1024) + fused residual into d_out ([B,C,H,W,L] fp32)
  qkv_gemm_kernel<<<12288, 128, 0, stream>>>(tok0, w_si, spa_in_b, Qb, Kb, Vt, 10);
  flash_attn_kernel<<<4096, 128, 0, stream>>>(Qb, Kb, Vt, attn, 10);
  out_gemm_kernel<<<4096, 128, 0, stream>>>(attn, w_so, spa_out_b, nullptr, x, outp, 1);
}
